// LlamaAttention_25958782337862
// MI455X (gfx1250) — matrix operand order repaired and checked by reading, not compiled, not run on hardware
//
#include <hip/hip_runtime.h>
#include <math.h>

#define DI __device__ __forceinline__

typedef __bf16 bf16;
typedef __attribute__((ext_vector_type(8)))  __bf16 v8bf;
typedef __attribute__((ext_vector_type(16))) __bf16 v16bf;
typedef __attribute__((ext_vector_type(8)))  float  v8f;

// ---------------- problem constants ----------------
constexpr int Bc   = 2;
constexpr int Sc   = 2048;
constexpr int Hc   = 4096;
constexpr int NHc  = 32;
constexpr int NKVc = 8;
constexpr int HDc  = 128;
constexpr int BS   = Bc * Sc;                    // 4096 rows
constexpr int QKVN = NHc*HDc + 2*NKVc*HDc;       // 6144 (q|k|v packed cols)

// ---------------- small helpers ----------------
union V16U { v16bf v; v8bf h[2]; };
DI v16bf mk16(v8bf lo, v8bf hi){ V16U u; u.h[0]=lo; u.h[1]=hi; return u.v; }

DI v8f v8f_zero(){
  v8f z;
  #pragma unroll
  for (int i=0;i<8;i++) z[i]=0.0f;
  return z;
}

DI v8f wmma_bf16(v16bf a, v16bf b, v8f c){
  // (neg_a, A, neg_b, B, c_mod, C, reuse_a, reuse_b)
  return __builtin_amdgcn_wmma_f32_16x16x32_bf16(false, a, false, b, (short)0, c, false, false);
}

// Flat shared-aperture address: addr[31:0] is the LDS byte offset (ISA 10.2)
DI unsigned lds_addr(const void* p){ return (unsigned)(uintptr_t)p; }

// gfx1250 async DMA global -> LDS, tracked by ASYNCcnt (ISA 15.18.3 op 98)
DI void async_b128(unsigned lds_byte_addr, const void* gaddr){
  asm volatile("global_load_async_to_lds_b128 %0, %1, off"
               :: "v"(lds_byte_addr), "v"(gaddr) : "memory");
}
DI void wait_async0(){ asm volatile("s_wait_asynccnt 0x0" ::: "memory"); }
DI void wait_async4(){ asm volatile("s_wait_asynccnt 0x4" ::: "memory"); }

// ---------------- fp32 -> bf16 convert ----------------
__global__ void cvt_bf16(const float* __restrict__ in, bf16* __restrict__ out, int n){
  int i = (blockIdx.x*blockDim.x + threadIdx.x)*4;
  if (i >= n) return;
  float4 v = *(const float4*)(in + i);
  out[i+0]=(bf16)v.x; out[i+1]=(bf16)v.y; out[i+2]=(bf16)v.z; out[i+3]=(bf16)v.w;
}

// fp32 [K][N] row-major  ->  bf16 [colOfs+N][K] (transposed, row stride K)
__global__ void cvt_transpose(const float* __restrict__ in, bf16* __restrict__ out,
                              int K, int N, long colOfs){
  long idx = (long)blockIdx.x*blockDim.x + threadIdx.x;
  if (idx >= (long)K*N) return;
  int  k = (int)(idx % K);
  long n = idx / K;
  out[(colOfs + n)*(long)K + k] = (bf16)in[(long)k*N + n];
}

// ---------------- bf16 GEMM: C[M,N] = A[M,K] * Bt[N,K]^T ----------------
// 128x128 block tile, 8 waves (wave32), wave = 32x64 sub-tile (2x4 WMMA tiles).
// Double-buffered LDS fed by async global->LDS DMA (ASYNCcnt pipelining).
template<bool F32OUT>
__global__ __launch_bounds__(256)
void gemm_bt(const bf16* __restrict__ A, const bf16* __restrict__ Bt,
             void* __restrict__ Cout, int K, int lda, int ldb, int ldc)
{
  constexpr int LDSL = 40;                    // 32 + 8 pad (keeps 16B alignment)
  __shared__ alignas(16) bf16 As[2][128*LDSL];
  __shared__ alignas(16) bf16 Bs[2][128*LDSL];

  const int tid  = threadIdx.x;
  const int lane = tid & 31;
  const int wave = tid >> 5;
  const int half = lane >> 4;
  const int l15  = lane & 15;
  const int wm   = wave & 3;                  // 4 waves along M
  const int wn   = wave >> 2;                 // 2 waves along N
  const size_t m0 = (size_t)blockIdx.y * 128;
  const size_t n0 = (size_t)blockIdx.x * 128;

  v8f acc[2][4];
  #pragma unroll
  for (int i=0;i<2;i++)
    #pragma unroll
    for (int j=0;j<4;j++) acc[i][j] = v8f_zero();

  const int lrow = tid >> 1;                  // 0..127
  const int lkh  = (tid & 1) * 16;            // 0 or 16

  unsigned ldsA[2], ldsB[2];
  #pragma unroll
  for (int p=0;p<2;p++){
    ldsA[p] = lds_addr(&As[p][lrow*LDSL + lkh]);
    ldsB[p] = lds_addr(&Bs[p][lrow*LDSL + lkh]);
  }
  const bf16* gA = A  + (m0 + lrow)*(size_t)lda + lkh;
  const bf16* gB = Bt + (n0 + lrow)*(size_t)ldb + lkh;

  // preload tile 0 into buffer 0 (4 async b128 per thread)
  async_b128(ldsA[0],      gA);
  async_b128(ldsA[0] + 16, gA + 8);
  async_b128(ldsB[0],      gB);
  async_b128(ldsB[0] + 16, gB + 8);

  const int nkt = K / 32;
  for (int kt = 0; kt < nkt; ++kt){
    const int cur = kt & 1;
    const int nxt = cur ^ 1;
    if (kt + 1 < nkt){
      const bf16* ga = gA + (size_t)(kt+1)*32;
      const bf16* gb = gB + (size_t)(kt+1)*32;
      async_b128(ldsA[nxt],      ga);
      async_b128(ldsA[nxt] + 16, ga + 8);
      async_b128(ldsB[nxt],      gb);
      async_b128(ldsB[nxt] + 16, gb + 8);
      wait_async4();                          // tile kt resident (in-order)
    } else {
      wait_async0();
    }
    __syncthreads();

    // A fragment: lane<16 -> K 0..7 & 16..23 ; lane>=16 -> K 8..15 & 24..31
    v16bf af[2];
    #pragma unroll
    for (int i=0;i<2;i++){
      const bf16* p = &As[cur][(wm*32 + i*16 + l15)*LDSL + half*8];
      af[i] = mk16(*(const v8bf*)p, *(const v8bf*)(p + 16));
    }
    // B fragment: lane<16 -> K 0..7 & 16..23 ; lane>=16 -> K 8..15 & 24..31, column = l15
    #pragma unroll
    for (int j=0;j<4;j++){
      const bf16* p = &Bs[cur][(wn*64 + j*16 + l15)*LDSL + half*8];
      v16bf bf = mk16(*(const v8bf*)p, *(const v8bf*)(p + 16));
      #pragma unroll
      for (int i=0;i<2;i++)
        acc[i][j] = wmma_bf16(af[i], bf, acc[i][j]);
    }
    __syncthreads();
  }

  // epilogue: C layout VGPR r -> M = r + 8*half, N = l15
  #pragma unroll
  for (int i=0;i<2;i++){
    #pragma unroll
    for (int j=0;j<4;j++){
      size_t mr = m0 + wm*32 + i*16 + half*8;
      size_t nc = n0 + wn*64 + j*16 + l15;
      #pragma unroll
      for (int r=0;r<8;r++){
        float v = acc[i][j][r];
        if constexpr (F32OUT) ((float*)Cout)[(mr+r)*(size_t)ldc + nc] = v;
        else                  ((bf16 *)Cout)[(mr+r)*(size_t)ldc + nc] = (bf16)v;
      }
    }
  }
}

// ---------------- RoPE on packed bf16 QKV (in place on q,k heads) ----------------
__global__ void rope_kernel(bf16* __restrict__ qkv,
                            const float* __restrict__ cosb,
                            const float* __restrict__ sinb)
{
  int idx = blockIdx.x*blockDim.x + threadIdx.x;
  if (idx >= BS*(NHc+NKVc)*(HDc/2)) return;
  int d    = idx & 63;
  int t    = idx >> 6;
  int head = t % (NHc + NKVc);
  int row  = t / (NHc + NKVc);                // b*S + s
  int col  = (head < NHc) ? head*HDc : (NHc*HDc + (head-NHc)*HDc);
  bf16* p = qkv + (size_t)row*QKVN + col;
  float x1 = (float)p[d], x2 = (float)p[d+64];
  const float* cr = cosb + (size_t)row*HDc;
  const float* sr = sinb + (size_t)row*HDc;
  p[d]    = (bf16)(x1*cr[d]    - x2*sr[d]);
  p[d+64] = (bf16)(x2*cr[d+64] + x1*sr[d+64]);
}

// ---------------- flash attention ----------------
// grid: (S/128 query blocks, B*NH).  8 waves; wave owns 16 query rows.
// Q fragments live in registers; K/V tiles of 64 keys staged in LDS.
__global__ __launch_bounds__(256)
void attn_fa(const bf16* __restrict__ qkv, bf16* __restrict__ attnout)
{
  constexpr int LDQ = 136;                       // 128 + 8 pad
  constexpr int LDP = 72;                        // 64 + 8 pad
  constexpr int KS_OFF = 0;                      // Ks: 64 x LDQ
  constexpr int VS_OFF = 64*LDQ;                 // Vs: 128 x LDP  ([d][key])
  constexpr int PS_OFF = VS_OFF + 128*LDP;       // Ps: 128 x LDP  ([row][key])
  __shared__ alignas(16) bf16 smem[PS_OFF + 128*LDP]; // 27136 elems = 54,272 B

  const int tid  = threadIdx.x;
  const int lane = tid & 31;
  const int wave = tid >> 5;
  const int half = lane >> 4;
  const int l15  = lane & 15;

  const int qblk = blockIdx.x;                   // 0..15
  const int bh   = blockIdx.y;                   // 0..63
  const int b    = bh / NHc;
  const int h    = bh % NHc;
  const int kvh  = h / (NHc / NKVc);

  const size_t rowQ0 = (size_t)(b*Sc + qblk*128);
  const int qcol = h*HDc;
  const int kcol = NHc*HDc + kvh*HDc;
  const int vcol = NHc*HDc + NKVc*HDc + kvh*HDc;
  const float scale = 0.08838834764831845f;      // HD^-0.5

  // ---- stage Q tile (128x128) via async DMA, then pull A-frags to registers
  {
    int row = tid >> 1;
    int c0  = (tid & 1) * 64;
    const bf16* g = qkv + (rowQ0 + row)*(size_t)QKVN + qcol + c0;
    unsigned s = lds_addr(smem + row*LDQ + c0);
    #pragma unroll
    for (int i=0;i<8;i++) async_b128(s + i*16, g + i*8);
    wait_async0();
  }
  __syncthreads();
  v16bf qfrag[4];
  {
    int row = wave*16 + l15;
    #pragma unroll
    for (int kk=0;kk<4;kk++){
      const bf16* p = smem + row*LDQ + kk*32 + half*8;
      qfrag[kk] = mk16(*(const v8bf*)p, *(const v8bf*)(p + 16));
    }
  }
  __syncthreads();

  v8f acc_o[8];
  #pragma unroll
  for (int j=0;j<8;j++) acc_o[j] = v8f_zero();
  float m_i[8], l_i[8];
  #pragma unroll
  for (int r=0;r<8;r++){ m_i[r] = -1e30f; l_i[r] = 0.0f; }

  bf16* Ks = smem + KS_OFF;
  bf16* Vs = smem + VS_OFF;
  bf16* Ps = smem + PS_OFF;

  const int nkt = 2*qblk + 2;                    // causal bound (64-key tiles)
  for (int kt = 0; kt < nkt; ++kt){
    // K tile (64x128) row-major via async DMA; V transposed into Vs[d][key]
    {
      int key = tid >> 2;                        // 0..63
      int d0  = (tid & 3) * 32;
      const size_t krow = (size_t)(b*Sc + kt*64 + key)*QKVN;
      const bf16* gk = qkv + krow + kcol + d0;
      unsigned sk = lds_addr(Ks + key*LDQ + d0);
      #pragma unroll
      for (int i=0;i<4;i++) async_b128(sk + i*16, gk + i*8);
      const bf16* gv = qkv + krow + vcol + d0;
      #pragma unroll
      for (int i=0;i<32;i++) Vs[(d0 + i)*LDP + key] = gv[i];
      if (kt + 1 < nkt) {                        // gfx1250 global_prefetch_b8
        __builtin_prefetch(gk + (size_t)64*QKVN, 0, 1);
        __builtin_prefetch(gv + (size_t)64*QKVN, 0, 1);
      }
      wait_async0();
    }
    __syncthreads();

    // S = Q * K^T  (wave: 16 rows x 64 keys)
    v8f acc_s[4];
    #pragma unroll
    for (int j=0;j<4;j++) acc_s[j] = v8f_zero();
    #pragma unroll
    for (int kk=0;kk<4;kk++){
      #pragma unroll
      for (int j=0;j<4;j++){
        const bf16* p = Ks + (j*16 + l15)*LDQ + kk*32 + half*8;
        v16bf bf = mk16(*(const v8bf*)p, *(const v8bf*)(p + 16));
        acc_s[j] = wmma_bf16(qfrag[kk], bf, acc_s[j]);
      }
    }

    // scale + causal mask
    const int qrow_base = qblk*128 + wave*16 + half*8;
    const bool need_mask = (kt >= 2*qblk);
    #pragma unroll
    for (int j=0;j<4;j++){
      int key = kt*64 + j*16 + l15;
      #pragma unroll
      for (int r=0;r<8;r++){
        float s = acc_s[j][r] * scale;
        if (need_mask && key > (qrow_base + r)) s = -1e30f;
        acc_s[j][r] = s;
      }
    }

    // online softmax (rows live in 16-lane halves -> shfl_xor 8..1)
    #pragma unroll
    for (int r=0;r<8;r++){
      float mx = -1e30f;
      #pragma unroll
      for (int j=0;j<4;j++) mx = fmaxf(mx, acc_s[j][r]);
      #pragma unroll
      for (int off=8; off>=1; off>>=1) mx = fmaxf(mx, __shfl_xor(mx, off, 32));
      float mnew  = fmaxf(m_i[r], mx);
      float alpha = __expf(m_i[r] - mnew);
      m_i[r] = mnew;
      float rsum = 0.0f;
      #pragma unroll
      for (int j=0;j<4;j++){
        float p = __expf(acc_s[j][r] - mnew);
        Ps[(wave*16 + half*8 + r)*LDP + j*16 + l15] = (bf16)p;
        rsum += p;
      }
      #pragma unroll
      for (int off=8; off>=1; off>>=1) rsum += __shfl_xor(rsum, off, 32);
      l_i[r] = l_i[r]*alpha + rsum;
      #pragma unroll
      for (int j=0;j<8;j++) acc_o[j][r] *= alpha;
    }

    // O += P * V   (P 16x64 from LDS as A-frags; intra-wave DS ordering)
    #pragma unroll
    for (int kk=0;kk<2;kk++){
      const bf16* pp = Ps + (wave*16 + l15)*LDP + kk*32 + half*8;
      v16bf pf = mk16(*(const v8bf*)pp, *(const v8bf*)(pp + 16));
      #pragma unroll
      for (int j=0;j<8;j++){
        const bf16* vp = Vs + (j*16 + l15)*LDP + kk*32 + half*8;
        v16bf bf = mk16(*(const v8bf*)vp, *(const v8bf*)(vp + 16));
        acc_o[j] = wmma_bf16(pf, bf, acc_o[j]);
      }
    }
    __syncthreads();
  }

  // normalize + write bf16 to attn buffer [BS, NH*HD]
  #pragma unroll
  for (int r=0;r<8;r++){
    float inv = (l_i[r] > 0.0f) ? (1.0f / l_i[r]) : 0.0f;
    size_t orow = rowQ0 + wave*16 + half*8 + r;
    #pragma unroll
    for (int j=0;j<8;j++)
      attnout[orow*(size_t)(NHc*HDc) + h*HDc + j*16 + l15] = (bf16)(acc_o[j][r] * inv);
  }
}

// ---------------- host launcher ----------------
extern "C" void kernel_launch(void* const* d_in, const int* in_sizes, int n_in,
                              void* d_out, int out_size, void* d_ws, size_t ws_size,
                              hipStream_t stream)
{
  const float* x    = (const float*)d_in[0];   // [B,S,H]
  const float* cosb = (const float*)d_in[1];   // [B,S,HD]
  const float* sinb = (const float*)d_in[2];   // [B,S,HD]
  const float* wq   = (const float*)d_in[3];   // [H, NH*HD]
  const float* wk   = (const float*)d_in[4];   // [H, NKV*HD]
  const float* wv   = (const float*)d_in[5];   // [H, NKV*HD]
  const float* wo   = (const float*)d_in[6];   // [NH*HD, H]
  float* out = (float*)d_out;

  // workspace layout (bf16): ~192 MiB total
  bf16* ws    = (bf16*)d_ws;
  bf16* Xbf   = ws;                             // 4096*4096
  bf16* Wtqkv = Xbf   + (size_t)BS*Hc;          // 6144*4096 (transposed, packed q|k|v)
  bf16* Wot   = Wtqkv + (size_t)QKVN*Hc;        // 4096*4096 (wo transposed)
  bf16* QKV   = Wot   + (size_t)Hc*Hc;          // 4096*6144
  bf16* Attn  = QKV   + (size_t)BS*QKVN;        // 4096*4096

  // convert / transpose weights & activations
  cvt_bf16<<<(BS*Hc/4 + 255)/256, 256, 0, stream>>>(x, Xbf, BS*Hc);
  cvt_transpose<<<(int)(((long)Hc*(NHc*HDc)  + 255)/256), 256, 0, stream>>>(wq, Wtqkv, Hc, NHc*HDc, 0);
  cvt_transpose<<<(int)(((long)Hc*(NKVc*HDc) + 255)/256), 256, 0, stream>>>(wk, Wtqkv, Hc, NKVc*HDc, NHc*HDc);
  cvt_transpose<<<(int)(((long)Hc*(NKVc*HDc) + 255)/256), 256, 0, stream>>>(wv, Wtqkv, Hc, NKVc*HDc, NHc*HDc + NKVc*HDc);
  cvt_transpose<<<(int)(((long)Hc*Hc         + 255)/256), 256, 0, stream>>>(wo, Wot, Hc, Hc, 0);

  // fused QKV projection: [4096,4096] x [4096,6144] -> bf16 [4096,6144]
  gemm_bt<false><<<dim3(QKVN/128, BS/128), 256, 0, stream>>>(Xbf, Wtqkv, QKV, Hc, Hc, Hc, QKVN);

  // RoPE on q and k heads
  int rope_total = BS * (NHc + NKVc) * (HDc/2);
  rope_kernel<<<(rope_total + 255)/256, 256, 0, stream>>>(QKV, cosb, sinb);

  // causal flash attention (GQA via kvh indexing)
  attn_fa<<<dim3(Sc/128, Bc*NHc), 256, 0, stream>>>(QKV, Attn);

  // output projection -> f32 d_out: [4096,4096] x [4096,4096]
  gemm_bt<true><<<dim3(Hc/128, BS/128), 256, 0, stream>>>(Attn, Wot, out, Hc, Hc, Hc, Hc);
}
